// MyLSTM_37495064494170
// MI455X (gfx1250) — matrix operand order repaired and checked by reading, not compiled, not run on hardware
//
#include <hip/hip_runtime.h>

// Problem constants (match reference)
#define B_   32
#define T_   128
#define E_   512
#define U_   1024
#define G4_  4096   // 4*U

typedef __attribute__((ext_vector_type(16))) __bf16 v16bf;
typedef __attribute__((ext_vector_type(8)))  __bf16 v8bf;
typedef __attribute__((ext_vector_type(8)))  float  v8f;
typedef __attribute__((ext_vector_type(4)))  int    v4i;

#define AS1 __attribute__((address_space(1)))
#define AS3 __attribute__((address_space(3)))

#if __has_builtin(__builtin_amdgcn_global_load_async_to_lds_b128)
#define USE_ASYNC_LDS 1
#else
#define USE_ASYNC_LDS 0
#endif

// Flat->LDS address: on CDNA5 the low 32 bits of a flat LDS address are the
// LDS byte offset (aperture rule), so truncation yields a valid AS(3) pointer.
__device__ __forceinline__ AS3 v4i* lds_v4(const void* p) {
  return (AS3 v4i*)(unsigned)(size_t)p;
}
__device__ __forceinline__ AS1 v4i* glb_v4(const void* p) {
  return (AS1 v4i*)(size_t)p;
}

__device__ __forceinline__ float sigm(float x) { return 1.0f / (1.0f + __expf(-x)); }

// ------------------------------------------------------------------
// Phase 0: state init, embedding gather (-> bf16), weight transpose
// ------------------------------------------------------------------
__global__ void k_zero_state(float* h, float* c, __bf16* hbf) {
  int i = blockIdx.x * blockDim.x + threadIdx.x;
  if (i < B_ * U_) { h[i] = 0.f; c[i] = 0.f; hbf[i] = (__bf16)0.f; }
}

__global__ void k_gather_embed(const int* __restrict__ tokens,
                               const float* __restrict__ emb,
                               __bf16* __restrict__ xbf) {
  // xbf[(t*B + b)*E + e] = emb[tokens[b*T + t]*E + e]
  size_t i = (size_t)blockIdx.x * blockDim.x + threadIdx.x;
  if (i >= (size_t)T_ * B_ * E_) return;
  int e   = (int)(i & (E_ - 1));
  int row = (int)(i >> 9);          // / E_
  int b   = row & (B_ - 1);
  int t   = row >> 5;               // / B_
  int tok = tokens[b * T_ + t];
  xbf[i] = (__bf16)emb[(size_t)tok * E_ + e];
}

__global__ void k_transpose_f32_bf16(const float* __restrict__ src, // [R][C]
                                     __bf16* __restrict__ dst,      // [C][R]
                                     int R, int C) {
  size_t i = (size_t)blockIdx.x * blockDim.x + threadIdx.x;
  if (i >= (size_t)R * C) return;
  int r = (int)(i / C);
  int c = (int)(i - (size_t)r * C);
  dst[(size_t)c * R + r] = (__bf16)src[i];
}

// ------------------------------------------------------------------
// Phase 1: XW[t*B+b][4U] = x @ W0 + b0 (one big WMMA GEMM, M=4096)
// A: bf16 [M][K] row-major.  Bt: bf16 [N][K] (B pre-transposed).
// C layout per CDNA5 WMMA f32 16x16 accumulator spec.
// ------------------------------------------------------------------
__global__ void k_gemm_xw(const __bf16* __restrict__ A,   // [4096][512]
                          const __bf16* __restrict__ Bt,  // [4096][512]
                          const float*  __restrict__ bias,
                          float* __restrict__ C,          // [4096][4096]
                          int K, int N) {
  int wave = threadIdx.x >> 5;
  int lane = threadIdx.x & 31;
  int m_tile = blockIdx.x;
  int n_tile = blockIdx.y * 8 + wave;
  int ln = lane & 15;
  int kh = lane >> 4;   // 0: lanes 0-15, 1: lanes 16-31

  const __bf16* arow = A  + (size_t)(m_tile * 16 + ln) * K + kh * 8;
  const __bf16* brow = Bt + (size_t)(n_tile * 16 + ln) * K + kh * 8;

  v8f acc = {};
  for (int k0 = 0; k0 < K; k0 += 32) {
    v8bf blo = *reinterpret_cast<const v8bf*>(brow + k0);
    v8bf bhi = *reinterpret_cast<const v8bf*>(brow + k0 + 16);
    v8bf alo = *reinterpret_cast<const v8bf*>(arow + k0);
    v8bf ahi = *reinterpret_cast<const v8bf*>(arow + k0 + 16);
    v16bf a, b;
#pragma unroll
    for (int j = 0; j < 8; ++j) { a[j] = alo[j]; a[8 + j] = ahi[j]; b[j] = blo[j]; b[8 + j] = bhi[j]; }
    acc = __builtin_amdgcn_wmma_f32_16x16x32_bf16(false, a, false, b,
                                                  (short)0, acc, false, false);
  }
  int col = n_tile * 16 + ln;
  float bv = bias[col];
#pragma unroll
  for (int r = 0; r < 8; ++r)
    C[(size_t)(m_tile * 16 + r + 8 * kh) * N + col] = acc[r] + bv;
}

// ------------------------------------------------------------------
// Recurrent GEMM: C[32][N] = A[32][K] @ Bt[N][K]^T.
// A staged in LDS via GLOBAL_LOAD_ASYNC_TO_LDS_B128 (ASYNCcnt path, no
// VGPR round-trip), row stride padded +8 bf16 to dodge bank conflicts.
// Each wave owns one 16-column tile and both 16-row tiles (M=32).
// ------------------------------------------------------------------
__global__ void k_gemm32(const __bf16* __restrict__ A,   // [32][K]
                         const __bf16* __restrict__ Bt,  // [N][K]
                         float* __restrict__ C,          // [32][N]
                         int K, int N) {
  extern __shared__ __bf16 sA[];          // 32 * (K+8) bf16
  const int pitch = K + 8;
  int tid = threadIdx.x;

  // Stage A -> LDS in 16-byte chunks
  int kchunks = K >> 3;
#if USE_ASYNC_LDS
  for (int i = tid; i < 32 * kchunks; i += blockDim.x) {
    int row = i / kchunks;
    int chk = i - row * kchunks;
    __builtin_amdgcn_global_load_async_to_lds_b128(
        glb_v4(reinterpret_cast<const uint4*>(A) + i),
        lds_v4(sA + (size_t)row * pitch + chk * 8),
        /*offset=*/0, /*cpol=*/0);
  }
#if __has_builtin(__builtin_amdgcn_s_wait_asynccnt)
  __builtin_amdgcn_s_wait_asynccnt(0);
#else
  asm volatile("s_wait_asynccnt 0" ::: "memory");
#endif
#else
  for (int i = tid; i < 32 * kchunks; i += blockDim.x) {
    int row = i / kchunks;
    int chk = i - row * kchunks;
    *reinterpret_cast<uint4*>(sA + (size_t)row * pitch + chk * 8) =
        reinterpret_cast<const uint4*>(A)[i];
  }
#endif
  __syncthreads();

  int wave = tid >> 5;
  int lane = tid & 31;
  int n_tile = blockIdx.x * 8 + wave;
  if (n_tile * 16 >= N) return;
  int ln = lane & 15;
  int kh = lane >> 4;

  const __bf16* brow = Bt + (size_t)(n_tile * 16 + ln) * K + kh * 8;
  const __bf16* a0p = sA + (size_t)ln * pitch + kh * 8;         // rows 0..15
  const __bf16* a1p = a0p + (size_t)16 * pitch;                 // rows 16..31

  v8f acc0 = {};
  v8f acc1 = {};
  for (int k0 = 0; k0 < K; k0 += 32) {
    __builtin_prefetch(brow + k0 + 64, 0, 0);   // global_prefetch_b8
    v8bf blo = *reinterpret_cast<const v8bf*>(brow + k0);
    v8bf bhi = *reinterpret_cast<const v8bf*>(brow + k0 + 16);
    v8bf a0lo = *reinterpret_cast<const v8bf*>(a0p + k0);
    v8bf a0hi = *reinterpret_cast<const v8bf*>(a0p + k0 + 16);
    v8bf a1lo = *reinterpret_cast<const v8bf*>(a1p + k0);
    v8bf a1hi = *reinterpret_cast<const v8bf*>(a1p + k0 + 16);
    v16bf a0, a1, b;
#pragma unroll
    for (int j = 0; j < 8; ++j) {
      a0[j] = a0lo[j]; a0[8 + j] = a0hi[j];
      a1[j] = a1lo[j]; a1[8 + j] = a1hi[j];
      b[j] = blo[j]; b[8 + j] = bhi[j];
    }
    acc0 = __builtin_amdgcn_wmma_f32_16x16x32_bf16(false, a0, false, b,
                                                   (short)0, acc0, false, false);
    acc1 = __builtin_amdgcn_wmma_f32_16x16x32_bf16(false, a1, false, b,
                                                   (short)0, acc1, false, false);
  }
  int col = n_tile * 16 + ln;
#pragma unroll
  for (int r = 0; r < 8; ++r) {
    C[(size_t)(r + 8 * kh) * N + col]      = acc0[r];
    C[(size_t)(16 + r + 8 * kh) * N + col] = acc1[r];
  }
}

// ------------------------------------------------------------------
// Gate nonlinearities. Cell 0: z0 = XW[t] (has +b0) + h@U0 (Cbuf cols
// 0..4095). Keras order i,f,g,o. c0_new is NOT carried (reference quirk);
// only out0 feeds cell 1.
// ------------------------------------------------------------------
__global__ void k_ew_cell0(const float* __restrict__ Cbuf,  // [32][8192]
                           const float* __restrict__ XW,    // [T*32][4096]
                           const float* __restrict__ cprev, // [32][1024]
                           __bf16* __restrict__ out0bf,     // [32][1024]
                           int t) {
  int i = blockIdx.x * blockDim.x + threadIdx.x;
  if (i >= B_ * U_) return;
  int b = i >> 10;
  int u = i & (U_ - 1);
  const float* xw = XW + ((size_t)t * B_ + b) * G4_;
  const float* z  = Cbuf + (size_t)b * (2 * G4_);
  float zi = xw[u]          + z[u];
  float zf = xw[U_ + u]     + z[U_ + u];
  float zg = xw[2 * U_ + u] + z[2 * U_ + u];
  float zo = xw[3 * U_ + u] + z[3 * U_ + u];
  float c0 = sigm(zf) * cprev[i] + sigm(zi) * tanhf(zg);
  out0bf[i] = (__bf16)(sigm(zo) * tanhf(c0));
}

// Cell 1: z1 = out0@W1 (Z1) + h@U1 (Cbuf cols 4096..8191) + b1.
// Updates the shared carry (h, c) and the bf16 copy of h.
__global__ void k_ew_cell1(const float* __restrict__ Z1,    // [32][4096]
                           const float* __restrict__ Cbuf,  // [32][8192]
                           const float* __restrict__ b1,
                           float* __restrict__ c,
                           float* __restrict__ h,
                           __bf16* __restrict__ hbf) {
  int i = blockIdx.x * blockDim.x + threadIdx.x;
  if (i >= B_ * U_) return;
  int b = i >> 10;
  int u = i & (U_ - 1);
  const float* z1 = Z1 + (size_t)b * G4_;
  const float* zp = Cbuf + (size_t)b * (2 * G4_) + G4_;
  float zi = z1[u]          + zp[u]          + b1[u];
  float zf = z1[U_ + u]     + zp[U_ + u]     + b1[U_ + u];
  float zg = z1[2 * U_ + u] + zp[2 * U_ + u] + b1[2 * U_ + u];
  float zo = z1[3 * U_ + u] + zp[3 * U_ + u] + b1[3 * U_ + u];
  float cn = sigm(zf) * c[i] + sigm(zi) * tanhf(zg);
  float hn = sigm(zo) * tanhf(cn);
  c[i] = cn;
  h[i] = hn;
  hbf[i] = (__bf16)hn;
}

// ------------------------------------------------------------------
// Final: sigmoid(h_last @ fc_w + fc_b), one block per batch row.
// ------------------------------------------------------------------
__global__ void k_final(const float* __restrict__ h,
                        const float* __restrict__ fcw,
                        const float* __restrict__ fcb,
                        float* __restrict__ out) {
  __shared__ float red[128];
  int b = blockIdx.x;
  float s = 0.f;
  for (int u = threadIdx.x; u < U_; u += 128)
    s += h[(size_t)b * U_ + u] * fcw[u];
  red[threadIdx.x] = s;
  __syncthreads();
  for (int st = 64; st > 0; st >>= 1) {
    if (threadIdx.x < st) red[threadIdx.x] += red[threadIdx.x + st];
    __syncthreads();
  }
  if (threadIdx.x == 0) out[b] = sigm(red[0] + fcb[0]);
}

// ------------------------------------------------------------------
// Host launcher
// ------------------------------------------------------------------
extern "C" void kernel_launch(void* const* d_in, const int* in_sizes, int n_in,
                              void* d_out, int out_size, void* d_ws, size_t ws_size,
                              hipStream_t stream) {
  (void)in_sizes; (void)n_in; (void)out_size; (void)ws_size;
  const int*   tokens = (const int*)  d_in[0];
  const float* emb    = (const float*)d_in[1];
  const float* W0     = (const float*)d_in[2];
  const float* U0     = (const float*)d_in[3];
  const float* b0     = (const float*)d_in[4];
  const float* W1     = (const float*)d_in[5];
  const float* U1     = (const float*)d_in[6];
  const float* b1     = (const float*)d_in[7];
  const float* fcw    = (const float*)d_in[8];
  const float* fcb    = (const float*)d_in[9];
  float* out = (float*)d_out;

  // Workspace carve-out (256B aligned)
  char* ws = (char*)d_ws;
  size_t off = 0;
  auto carve = [&](size_t bytes) -> void* {
    void* p = ws + off;
    off = (off + bytes + 255) & ~(size_t)255;
    return p;
  };
  float*  XW     = (float*) carve((size_t)T_ * B_ * G4_ * sizeof(float)); // 64 MB
  __bf16* xbf    = (__bf16*)carve((size_t)T_ * B_ * E_ * sizeof(__bf16)); // 4 MB
  __bf16* U01t   = (__bf16*)carve((size_t)2 * G4_ * U_ * sizeof(__bf16)); // 16 MB
  __bf16* W1t    = (__bf16*)carve((size_t)G4_ * U_ * sizeof(__bf16));     // 8 MB
  __bf16* W0t    = (__bf16*)carve((size_t)G4_ * E_ * sizeof(__bf16));     // 4 MB
  float*  Cbuf   = (float*) carve((size_t)B_ * 2 * G4_ * sizeof(float));  // 1 MB
  float*  Z1     = (float*) carve((size_t)B_ * G4_ * sizeof(float));
  float*  hf     = (float*) carve((size_t)B_ * U_ * sizeof(float));
  float*  cf     = (float*) carve((size_t)B_ * U_ * sizeof(float));
  __bf16* hbf    = (__bf16*)carve((size_t)B_ * U_ * sizeof(__bf16));
  __bf16* out0bf = (__bf16*)carve((size_t)B_ * U_ * sizeof(__bf16));

  dim3 blk(256);

  // Phase 0
  k_zero_state<<<dim3((B_ * U_ + 255) / 256), blk, 0, stream>>>(hf, cf, hbf);
  k_gather_embed<<<dim3((T_ * B_ * E_ + 255) / 256), blk, 0, stream>>>(tokens, emb, xbf);
  k_transpose_f32_bf16<<<dim3((U_ * G4_ + 255) / 256), blk, 0, stream>>>(U0, U01t, U_, G4_);
  k_transpose_f32_bf16<<<dim3((U_ * G4_ + 255) / 256), blk, 0, stream>>>(U1, U01t + (size_t)G4_ * U_, U_, G4_);
  k_transpose_f32_bf16<<<dim3((U_ * G4_ + 255) / 256), blk, 0, stream>>>(W1, W1t, U_, G4_);
  k_transpose_f32_bf16<<<dim3((E_ * G4_ + 255) / 256), blk, 0, stream>>>(W0, W0t, E_, G4_);

  // Phase 1: XW = x @ W0 + b0 over all timesteps  (M=4096, K=512, N=4096)
  k_gemm_xw<<<dim3(256, 32), blk, 0, stream>>>(xbf, W0t, b0, XW, E_, G4_);

  // Phase 2: recurrence
  size_t ldsb = (size_t)32 * (U_ + 8) * sizeof(__bf16);
  for (int t = 0; t < T_; ++t) {
    // [h@U0 | h@U1] : M=32, K=1024, N=8192
    k_gemm32<<<dim3(64), blk, ldsb, stream>>>(hbf, U01t, Cbuf, U_, 2 * G4_);
    k_ew_cell0<<<dim3(128), blk, 0, stream>>>(Cbuf, XW, cf, out0bf, t);
    // out0 @ W1 : M=32, K=1024, N=4096
    k_gemm32<<<dim3(32), blk, ldsb, stream>>>(out0bf, W1t, Z1, U_, G4_);
    k_ew_cell1<<<dim3(128), blk, 0, stream>>>(Z1, Cbuf, b1, cf, hf, hbf);
  }

  // Phase 3
  k_final<<<dim3(B_), dim3(128), 0, stream>>>(hf, fcw, fcb, out);
}
